// MiddleHeadLayer_87385404604918
// MI455X (gfx1250) — compile-verified, matrix operand order checked by reading, not run on hardware
//
#include <hip/hip_runtime.h>
#include <math.h>

#define DM   1024
#define DFF  4096
#define NROW 16384

typedef __attribute__((ext_vector_type(16))) __bf16 v16bf;
typedef __attribute__((ext_vector_type(8)))  __bf16 v8bf;
typedef __attribute__((ext_vector_type(4)))  __bf16 v4bf;
typedef __attribute__((ext_vector_type(8)))  float  v8f;
typedef __attribute__((ext_vector_type(4)))  unsigned int u32x4;
typedef __attribute__((ext_vector_type(8)))  int          i32x8;
typedef __attribute__((ext_vector_type(4)))  int          i32x4;

#define LDS_STRIDE 40   // 32 + 8 pad (bf16); TDM pad_interval=64B / pad_amount=16B reproduces this

#if __has_builtin(__builtin_amdgcn_tensor_load_to_lds)
#define USE_TDM 1
#else
#define USE_TDM 0
#endif

// LDS byte offset of a generic pointer into shared memory (addrspacecast to AS3).
// Body is device-only: the host pass parses __device__ functions and would reject
// the 64-bit->32-bit pointer cast (AS3 pointers are 32-bit only on the device).
__device__ __forceinline__ unsigned lds_byte(const void* p) {
#if defined(__HIP_DEVICE_COMPILE__)
  return (unsigned)(unsigned long long)(__attribute__((address_space(3))) const void*)p;
#else
  return 0;
#endif
}

#if USE_TDM
// Issue one TDM load of a 128-row x 32-col bf16 tile (row stride K elems) into LDS,
// with 16B of LDS padding inserted after every 64B row (=> LDS_STRIDE 40 halfwords).
__device__ __forceinline__ void tdm_load_tile(const __bf16* gptr, unsigned lds_addr, int K) {
  unsigned long long ga = (unsigned long long)gptr;
  u32x4 g0;
  g0[0] = 1u;                                   // count=1 (valid descriptor)
  g0[1] = lds_addr;                             // LDS byte address
  g0[2] = (unsigned)ga;                         // global_addr[31:0]
  g0[3] = (unsigned)((ga >> 32) & 0x1FFFFFFull) // global_addr[56:32]
        | (2u << 30);                           // type=2 ("image")
  i32x8 g1;
  g1[0] = (int)((1u << 16)    // data_size = 1 -> 2 bytes
              | (1u << 20)    // pad_enable
              | (3u << 22)    // pad_interval: code 3 -> every 8x8B = 64B
              | (3u << 25));  // pad_amount:   code 3 -> 4 DWORDs = 16B
  g1[1] = (int)((unsigned)(K & 0xFFFF) << 16);  // tensor_dim0[15:0] in bits 63:48
  g1[2] = 0;                                    // tensor_dim0[31:16]=0, tensor_dim1[15:0]=0
  g1[3] = (int)(0x10u | (32u << 16));           // tensor_dim1[31:16]=0x10 (1M rows), tile_dim0=32
  g1[4] = 128;                                  // tile_dim1 = 128 rows
  g1[5] = K;                                    // tensor_dim0_stride[31:0]
  g1[6] = 0;
  g1[7] = 0;
  i32x4 z4 = {0, 0, 0, 0};
#if __clang_major__ >= 23
  i32x8 z8 = {};
  __builtin_amdgcn_tensor_load_to_lds(g0, g1, z4, z4, z8, 0);
#else
  __builtin_amdgcn_tensor_load_to_lds(g0, g1, z4, z4, 0);
#endif
}
#endif

// --- fragment load matching CDNA5 16-bit A/B VGPR layout (ISA 7.12.2) ---
__device__ __forceinline__ v16bf load_frag_lds(const __bf16* base, int row, int khalf) {
  const __bf16* p = base + row * LDS_STRIDE + khalf * 8;
  v8bf lo = *(const v8bf*)(p);        // K = khalf*8 .. +7
  v8bf hi = *(const v8bf*)(p + 16);   // K = 16 + khalf*8 .. +7
  return __builtin_shufflevector(lo, hi, 0,1,2,3,4,5,6,7,8,9,10,11,12,13,14,15);
}

// Shared GEMM core: C_tile(128x128) += A[mBlk..][K] * B[nBlk..][K]^T, bf16 in, f32 acc.
// As/Bs are double-buffered: 2 x 128 x LDS_STRIDE halfwords each.
template<int KDIM>
__device__ __forceinline__ void gemm_tiles(const __bf16* __restrict__ A,
                                           const __bf16* __restrict__ B,
                                           int mBlk, int nBlk,
                                           __bf16* As, __bf16* Bs,
                                           int wm, int wn, int l16, int lhalf,
                                           v8f (&acc)[4][2]) {
  const int wave = threadIdx.x >> 5;
  constexpr int BUFE = 128 * LDS_STRIDE;   // halfwords per buffer

#if USE_TDM
  const unsigned asB = lds_byte(As), bsB = lds_byte(Bs);
  if (wave == 0) {   // one wave drives the Tensor Data Mover
    tdm_load_tile(A + (size_t)mBlk * KDIM, asB, KDIM);
    tdm_load_tile(B + (size_t)nBlk * KDIM, bsB, KDIM);
  }
  for (int k0 = 0; k0 < KDIM; k0 += 32) {
    const int cur = (k0 >> 5) & 1;
    if (wave == 0) __builtin_amdgcn_s_wait_tensorcnt(0);  // current buffer landed
    __syncthreads();
    if (wave == 0 && k0 + 32 < KDIM) {  // overlap next DMA with this tile's WMMAs
      tdm_load_tile(A + (size_t)mBlk * KDIM + (k0 + 32), asB + (cur ^ 1) * BUFE * 2, KDIM);
      tdm_load_tile(B + (size_t)nBlk * KDIM + (k0 + 32), bsB + (cur ^ 1) * BUFE * 2, KDIM);
    }
    const __bf16* Ac = As + cur * BUFE;
    const __bf16* Bc = Bs + cur * BUFE;
#else
  const int tid = threadIdx.x;
  const int r0 = (tid + 0)   >> 2, c0 = (tid + 0)   & 3;
  const int r1 = (tid + 256) >> 2, c1 = (tid + 256) & 3;
  for (int k0 = 0; k0 < KDIM; k0 += 32) {
    uint4 a0 = *(const uint4*)(A + (size_t)(mBlk + r0) * KDIM + k0 + c0 * 8);
    uint4 a1 = *(const uint4*)(A + (size_t)(mBlk + r1) * KDIM + k0 + c1 * 8);
    uint4 b0 = *(const uint4*)(B + (size_t)(nBlk + r0) * KDIM + k0 + c0 * 8);
    uint4 b1 = *(const uint4*)(B + (size_t)(nBlk + r1) * KDIM + k0 + c1 * 8);
    if (k0 + 32 < KDIM) {
      __builtin_prefetch(A + (size_t)(mBlk + r0) * KDIM + k0 + 32 + c0 * 8, 0, 1);
      __builtin_prefetch(B + (size_t)(nBlk + r0) * KDIM + k0 + 32 + c0 * 8, 0, 1);
    }
    __syncthreads();
    *(uint4*)(&As[r0 * LDS_STRIDE + c0 * 8]) = a0;
    *(uint4*)(&As[r1 * LDS_STRIDE + c1 * 8]) = a1;
    *(uint4*)(&Bs[r0 * LDS_STRIDE + c0 * 8]) = b0;
    *(uint4*)(&Bs[r1 * LDS_STRIDE + c1 * 8]) = b1;
    __syncthreads();
    const __bf16* Ac = As;
    const __bf16* Bc = Bs;
#endif

    v16bf af[4], bf[2];
#pragma unroll
    for (int mf = 0; mf < 4; ++mf)
      af[mf] = load_frag_lds(Ac, wm * 64 + mf * 16 + l16, lhalf);
#pragma unroll
    for (int nf = 0; nf < 2; ++nf)
      bf[nf] = load_frag_lds(Bc, wn * 32 + nf * 16 + l16, lhalf);
#pragma unroll
    for (int mf = 0; mf < 4; ++mf)
#pragma unroll
      for (int nf = 0; nf < 2; ++nf)
        acc[mf][nf] = __builtin_amdgcn_wmma_f32_16x16x32_bf16(
            false, af[mf], false, bf[nf], (short)0, acc[mf][nf], false, false);
  }
}

// Layer 1: inner = tanh(X @ W1^T + b1), stored bf16.
__global__ __launch_bounds__(256)
void gemm1_tanh(const __bf16* __restrict__ A,    // X bf16 [NROW, DM]
                const __bf16* __restrict__ B,    // W1 bf16 [DFF, DM]
                const float*  __restrict__ bias, // b1 [DFF]
                __bf16* __restrict__ C)          // inner bf16 [NROW, DFF]
{
  __shared__ __bf16 As[2 * 128 * LDS_STRIDE];
  __shared__ __bf16 Bs[2 * 128 * LDS_STRIDE];
  const int tid = threadIdx.x, wave = tid >> 5, lane = tid & 31;
  const int l16 = lane & 15, lhalf = lane >> 4;
  const int wm = wave >> 2, wn = wave & 3;
  const int mBlk = blockIdx.y * 128, nBlk = blockIdx.x * 128;

  v8f acc[4][2] = {};
  gemm_tiles<DM>(A, B, mBlk, nBlk, As, Bs, wm, wn, l16, lhalf, acc);

  // C layout: VGPR v -> row (v + 8*lhalf), col = l16 (ISA 7.12.2)
#pragma unroll
  for (int mf = 0; mf < 4; ++mf)
#pragma unroll
    for (int nf = 0; nf < 2; ++nf) {
      const int col = nBlk + wn * 32 + nf * 16 + l16;
      const float bv = bias[col];
#pragma unroll
      for (int v = 0; v < 8; ++v) {
        const int row = mBlk + wm * 64 + mf * 16 + lhalf * 8 + v;
        C[(size_t)row * DFF + col] = (__bf16)tanhf(acc[mf][nf][v] + bv);
      }
    }
}

// Layer 2 fused with the row-dot: partials[row] += sum_col tanh(inner@W2^T+b2)*X.
__global__ __launch_bounds__(256)
void gemm2_tanh_dot(const __bf16* __restrict__ A,    // inner bf16 [NROW, DFF]
                    const __bf16* __restrict__ B,    // W2 bf16 [DM, DFF]
                    const float*  __restrict__ bias, // b2 [DM]
                    const float*  __restrict__ X,    // batch f32 [NROW, DM]
                    float* __restrict__ partials)    // [NROW]
{
  __shared__ __bf16 As[2 * 128 * LDS_STRIDE];
  __shared__ __bf16 Bs[2 * 128 * LDS_STRIDE];
  const int tid = threadIdx.x, wave = tid >> 5, lane = tid & 31;
  const int l16 = lane & 15, lhalf = lane >> 4;
  const int wm = wave >> 2, wn = wave & 3;
  const int mBlk = blockIdx.y * 128, nBlk = blockIdx.x * 128;

  v8f acc[4][2] = {};
  gemm_tiles<DFF>(A, B, mBlk, nBlk, As, Bs, wm, wn, l16, lhalf, acc);

#pragma unroll
  for (int mf = 0; mf < 4; ++mf) {
#pragma unroll
    for (int v = 0; v < 8; ++v) {
      const int row = mBlk + wm * 64 + mf * 16 + lhalf * 8 + v;
      float s = 0.0f;
#pragma unroll
      for (int nf = 0; nf < 2; ++nf) {
        const int col = nBlk + wn * 32 + nf * 16 + l16;
        const float wx = tanhf(acc[mf][nf][v] + bias[col]);
        s += wx * X[(size_t)row * DM + col];
      }
      // reduce across the 16 N-lanes; xor masks < 16 keep the wave32 halves separate
      s += __shfl_xor(s, 1, 32);
      s += __shfl_xor(s, 2, 32);
      s += __shfl_xor(s, 4, 32);
      s += __shfl_xor(s, 8, 32);
      if (l16 == 0) atomicAdd(&partials[row], s);
    }
  }
}

__global__ void cvt_f32_bf16(const float* __restrict__ s, __bf16* __restrict__ d, int n4) {
  int i = blockIdx.x * blockDim.x + threadIdx.x;
  if (i < n4) {
    float4 f = ((const float4*)s)[i];
    v4bf o;
    o[0] = (__bf16)f.x; o[1] = (__bf16)f.y; o[2] = (__bf16)f.z; o[3] = (__bf16)f.w;
    ((v4bf*)d)[i] = o;
  }
}

__global__ void zero_f32(float* __restrict__ p, int n) {
  int i = blockIdx.x * blockDim.x + threadIdx.x;
  if (i < n) p[i] = 0.0f;
}

__global__ void sigmoid_k(const float* __restrict__ p, float* __restrict__ o, int n) {
  int i = blockIdx.x * blockDim.x + threadIdx.x;
  if (i < n) o[i] = 1.0f / (1.0f + __expf(-p[i]));
}

extern "C" void kernel_launch(void* const* d_in, const int* in_sizes, int n_in,
                              void* d_out, int out_size, void* d_ws, size_t ws_size,
                              hipStream_t stream) {
  const float* batch = (const float*)d_in[0];
  const float* W1    = (const float*)d_in[1];
  const float* b1    = (const float*)d_in[2];
  const float* W2    = (const float*)d_in[3];
  const float* b2    = (const float*)d_in[4];
  float* out = (float*)d_out;

  // Workspace layout (needs ~185 MB):
  //   X bf16 (32MB) | W1 bf16 (8MB) | W2 bf16 (8MB) | inner bf16 (128MB) | partials f32 (64KB)
  char* ws = (char*)d_ws;
  __bf16* batch_bf = (__bf16*)ws;
  __bf16* W1_bf    = (__bf16*)(ws + (size_t)NROW * DM * 2);
  __bf16* W2_bf    = (__bf16*)(ws + ((size_t)NROW * DM + (size_t)DFF * DM) * 2);
  __bf16* inner_bf = (__bf16*)(ws + ((size_t)NROW * DM + 2 * (size_t)DFF * DM) * 2);
  float*  partials = (float*)(ws + ((size_t)NROW * DM + 2 * (size_t)DFF * DM +
                                    (size_t)NROW * DFF) * 2);

  const int thr = 256;
  cvt_f32_bf16<<<(NROW * DM / 4 + thr - 1) / thr, thr, 0, stream>>>(batch, batch_bf, NROW * DM / 4);
  cvt_f32_bf16<<<(DFF  * DM / 4 + thr - 1) / thr, thr, 0, stream>>>(W1, W1_bf, DFF * DM / 4);
  cvt_f32_bf16<<<(DM  * DFF / 4 + thr - 1) / thr, thr, 0, stream>>>(W2, W2_bf, DM * DFF / 4);
  zero_f32<<<(NROW + thr - 1) / thr, thr, 0, stream>>>(partials, NROW);

  dim3 g1(DFF / 128, NROW / 128);   // 32 x 128 workgroups
  gemm1_tanh<<<g1, thr, 0, stream>>>(batch_bf, W1_bf, b1, inner_bf);

  dim3 g2(DM / 128, NROW / 128);    // 8 x 128 workgroups
  gemm2_tanh_dot<<<g2, thr, 0, stream>>>(inner_bf, W2_bf, b2, batch, partials);

  sigmoid_k<<<(NROW + thr - 1) / thr, thr, 0, stream>>>(partials, out, NROW);
}
